// GNNClassifier_88648124990421
// MI455X (gfx1250) — matrix operand order repaired and checked by reading, not compiled, not run on hardware
//
#include <hip/hip_runtime.h>
#include <hip/hip_fp16.h>

typedef __attribute__((ext_vector_type(16))) _Float16 v16h;
typedef __attribute__((ext_vector_type(8)))  _Float16 v8h;
typedef __attribute__((ext_vector_type(8)))  float    v8f;

#define THREADS 256
#define FDIM 128
#define NBLK 4   // N-tiles (16 cols each) per wave: 16x64 output per wave
__device__ __constant__ float kNegSlope = 0.2f;

// -------------------- small utility kernels --------------------

__global__ __launch_bounds__(THREADS)
void k_fill_f32(float* __restrict__ p, float v, long n) {
  long i = (long)blockIdx.x * blockDim.x + threadIdx.x;
  if (i < n) p[i] = v;
}

__global__ __launch_bounds__(THREADS)
void k_embed(const int* __restrict__ x, const float* __restrict__ emb,
             _Float16* __restrict__ h16, int N) {
  long i = (long)blockIdx.x * blockDim.x + threadIdx.x;  // over N*64
  if (i >= (long)N * 64) return;
  long n = i >> 6;
  int  k = (int)(i & 63);
  h16[i] = (_Float16)emb[(long)x[n] * 64 + k];
}

// Pack row-major f32 W[K x N] into WMMA-B tile layout:
// Bp[((kt*ntiles + nt)*32 + lane)*16 + j] = W[kt*32 + (lane>=16?16:0) + j][nt*16 + (lane&15)]
__global__ __launch_bounds__(THREADS)
void k_pack_b(const float* __restrict__ W, _Float16* __restrict__ Bp, int K, int N) {
  int idx = blockIdx.x * blockDim.x + threadIdx.x;
  if (idx >= K * N) return;
  int j    = idx & 15;
  int lane = (idx >> 4) & 31;
  int tile = idx >> 9;
  int ntiles = N >> 4;
  int kt = tile / ntiles;
  int nt = tile - kt * ntiles;
  int k = kt * 32 + ((lane & 16) ? 16 : 0) + j;
  int n = nt * 16 + (lane & 15);
  Bp[idx] = (_Float16)W[(long)k * N + n];
}

// -------------------- WMMA GEMM: C[MxN] = A[MxK](f16) * Bp(packed f16) ----------
// One wave computes a 16x(16*NBLK) output block: the A fragment is loaded once
// per K-tile and reused for NBLK back-to-back v_wmma issues.

__global__ __launch_bounds__(THREADS)
void k_gemm_wmma(const _Float16* __restrict__ A, const _Float16* __restrict__ Bp,
                 float* __restrict__ C, int M, int K, int N) {
  const int lane    = threadIdx.x & 31;
  const int wave    = (int)(((long)blockIdx.x * blockDim.x + threadIdx.x) >> 5);
  const int ntiles  = N >> 4;
  const int ngroups = ntiles / NBLK;
  const int mTile   = wave / ngroups;
  const int nGroup  = wave - mTile * ngroups;
  if (mTile * 16 >= M) return;               // wave-uniform exit (EXEC stays all-ones)

  const bool hi  = lane >= 16;
  int row = mTile * 16 + (lane & 15);
  if (row >= M) row = M - 1;                 // clamp (stores guarded below)
  const int klo = hi ? 16 : 0;

  v8f acc[NBLK];
#pragma unroll
  for (int t = 0; t < NBLK; ++t) acc[t] = (v8f){};

  const int ktiles = K >> 5;
  const _Float16* arow = A + (long)row * K + klo;
  const _Float16* bptr = Bp + ((long)(nGroup * NBLK) * 32 + lane) * 16;
  const long bstride   = (long)ntiles * 32 * 16;  // halfs per K-tile of packed B

  for (int kt = 0; kt < ktiles; ++kt) {
    v8h alo = *(const v8h*)(arow);
    v8h ahi = *(const v8h*)(arow + 8);
    v16h a;
#pragma unroll
    for (int j = 0; j < 8; ++j) { a[j] = alo[j]; a[j + 8] = ahi[j]; }
#pragma unroll
    for (int t = 0; t < NBLK; ++t) {
      v16h b = *(const v16h*)(bptr + (long)t * 32 * 16);
      acc[t] = __builtin_amdgcn_wmma_f32_16x16x32_f16(
          /*neg_a=*/false, a, /*neg_b=*/false, b,
          /*c_mod=*/(short)0, acc[t], /*reuse_a=*/false, /*reuse_b=*/false);
    }
    arow += 32;
    bptr += bstride;
    __builtin_prefetch(arow, 0, 1);          // global_prefetch_b8 (speculative)
  }

  const int colBase = nGroup * NBLK * 16 + (lane & 15);
#pragma unroll
  for (int r = 0; r < 8; ++r) {
    int m = mTile * 16 + r + (hi ? 8 : 0);
    if (m < M) {
      float* crow = C + (long)m * N + colBase;
#pragma unroll
      for (int t = 0; t < NBLK; ++t) crow[t * 16] = acc[t][r];
    }
  }
}

// -------------------- attention logits --------------------

__global__ __launch_bounds__(THREADS)
void k_alpha(const float* __restrict__ hg, const float* __restrict__ a_src,
             const float* __restrict__ a_dst, float* __restrict__ alS,
             float* __restrict__ alD, int N) {
  long i = (long)blockIdx.x * blockDim.x + threadIdx.x;  // N*4
  if (i >= (long)N * 4) return;
  long n = i >> 2;
  int  h = (int)(i & 3);
  const float* hp = hg + n * FDIM + h * 32;
  const float* as = a_src + h * 32;
  const float* ad = a_dst + h * 32;
  float ss = 0.f, sd = 0.f;
#pragma unroll 8
  for (int d = 0; d < 32; ++d) { float v = hp[d]; ss += v * as[d]; sd += v * ad[d]; }
  alS[i] = ss;
  alD[i] = sd;
}

// -------------------- edge passes --------------------

__device__ inline void atomicMaxFloat(float* addr, float value) {
  if (value >= 0.f) atomicMax((int*)addr, __float_as_int(value));
  else              atomicMin((unsigned int*)addr, (unsigned int)__float_as_int(value));
}

__global__ __launch_bounds__(THREADS)
void k_edge_max(const int* __restrict__ src, const int* __restrict__ dst,
                long E, long Et, const float* __restrict__ alS,
                const float* __restrict__ alD, float* __restrict__ e,
                float* __restrict__ mx) {
  long i = (long)blockIdx.x * blockDim.x + threadIdx.x;
  if (i >= Et) return;
  int s = (i < E) ? src[i] : (int)(i - E);
  int d = (i < E) ? dst[i] : (int)(i - E);
#pragma unroll
  for (int h = 0; h < 4; ++h) {
    float v = alS[(long)s * 4 + h] + alD[(long)d * 4 + h];
    v = (v > 0.f) ? v : kNegSlope * v;       // leaky relu
    e[i * 4 + h] = v;
    atomicMaxFloat(&mx[(long)d * 4 + h], v);
  }
}

__global__ __launch_bounds__(THREADS)
void k_edge_expsum(const int* __restrict__ src, const int* __restrict__ dst,
                   long E, long Et, const float* __restrict__ mx,
                   float* __restrict__ e, float* __restrict__ ssum) {
  long i = (long)blockIdx.x * blockDim.x + threadIdx.x;
  if (i >= Et) return;
  int d = (i < E) ? dst[i] : (int)(i - E);
#pragma unroll
  for (int h = 0; h < 4; ++h) {
    float ex = __expf(e[i * 4 + h] - mx[(long)d * 4 + h]);
    e[i * 4 + h] = ex;                        // store exp in place
    atomicAdd(&ssum[(long)d * 4 + h], ex);
  }
}

__global__ __launch_bounds__(THREADS)
void k_edge_agg(const int* __restrict__ src, const int* __restrict__ dst,
                long E, long Et, const float* __restrict__ e,
                const float* __restrict__ ssum, const float* __restrict__ hg,
                float* __restrict__ agg) {
  long t = (long)blockIdx.x * blockDim.x + threadIdx.x;  // Et*4
  if (t >= Et * 4) return;
  long i = t >> 2;
  int  h = (int)(t & 3);
  int s = (i < E) ? src[i] : (int)(i - E);
  int d = (i < E) ? dst[i] : (int)(i - E);
  float coef = e[t] / (ssum[(long)d * 4 + h] + 1e-16f);
  const float4* hp = (const float4*)(hg + (long)s * FDIM + h * 32);
  float*        op = agg + (long)d * FDIM + h * 32;
#pragma unroll
  for (int q = 0; q < 8; ++q) {
    float4 v = hp[q];
    atomicAdd(&op[q * 4 + 0], v.x * coef);
    atomicAdd(&op[q * 4 + 1], v.y * coef);
    atomicAdd(&op[q * 4 + 2], v.z * coef);
    atomicAdd(&op[q * 4 + 3], v.w * coef);
  }
}

// -------------------- epilogues --------------------

// hg <- relu(agg + b)  (in place over hg), h16 <- f16(hg) for next layer's GEMM
__global__ __launch_bounds__(THREADS)
void k_bias_relu(const float* __restrict__ agg, const float* __restrict__ b,
                 float* __restrict__ hg, _Float16* __restrict__ h16, int N) {
  long i = (long)blockIdx.x * blockDim.x + threadIdx.x;
  if (i >= (long)N * FDIM) return;
  int k = (int)(i & (FDIM - 1));
  float v = agg[i] + b[k];
  v = (v > 0.f) ? v : 0.f;
  hg[i]  = v;
  h16[i] = (_Float16)v;
}

__global__ __launch_bounds__(THREADS)
void k_pool(const float* __restrict__ h, const int* __restrict__ batch,
            float* __restrict__ pooled, float* __restrict__ cnt, int N) {
  long n = (long)blockIdx.x * blockDim.x + threadIdx.x;
  if (n >= N) return;
  int g = batch[n];
  atomicAdd(&cnt[g], 1.f);
  const float* hp = h + n * FDIM;
  float*       pp = pooled + (long)g * FDIM;
#pragma unroll 8
  for (int k = 0; k < FDIM; ++k) atomicAdd(&pp[k], hp[k]);
}

__global__ __launch_bounds__(THREADS)
void k_final(const float* __restrict__ pooled, const float* __restrict__ cnt,
             const float* __restrict__ linW, const float* __restrict__ linb,
             float* __restrict__ out, int G) {
  int i = blockIdx.x * blockDim.x + threadIdx.x;  // G*2
  if (i >= G * 2) return;
  int g = i >> 1, c = i & 1;
  float inv = 1.f / fmaxf(cnt[g], 1.f);
  const float* pp = pooled + (long)g * FDIM;
  float s = 0.f;
#pragma unroll 8
  for (int k = 0; k < FDIM; ++k) s += pp[k] * inv * linW[k * 2 + c];
  out[i] = s + linb[c];
}

// -------------------- host launcher --------------------

extern "C" void kernel_launch(void* const* d_in, const int* in_sizes, int n_in,
                              void* d_out, int out_size, void* d_ws, size_t ws_size,
                              hipStream_t stream) {
  const int*   x     = (const int*)d_in[0];
  const int*   ei    = (const int*)d_in[1];
  const int*   batch = (const int*)d_in[2];
  const float* emb   = (const float*)d_in[3];
  const float* W1    = (const float*)d_in[4];
  const float* as1   = (const float*)d_in[5];
  const float* ad1   = (const float*)d_in[6];
  const float* b1    = (const float*)d_in[7];
  const float* W2    = (const float*)d_in[8];
  const float* as2   = (const float*)d_in[9];
  const float* ad2   = (const float*)d_in[10];
  const float* b2    = (const float*)d_in[11];
  const float* linW  = (const float*)d_in[12];
  const float* linb  = (const float*)d_in[13];
  float*       out   = (float*)d_out;

  const long N  = in_sizes[0];
  const long E  = in_sizes[1] / 2;
  const long Et = E + N;
  const int  G  = out_size / 2;
  const int* srcIdx = ei;
  const int* dstIdx = ei + E;

  // carve workspace (256B-aligned sections)
  char* w = (char*)d_ws;
  auto take = [&](size_t bytes) -> char* {
    char* p = w;
    w += (bytes + 255) & ~(size_t)255;
    return p;
  };
  _Float16* h16    = (_Float16*)take((size_t)N * FDIM * 2);
  float*    hg     = (float*)take((size_t)N * FDIM * 4);
  float*    agg    = (float*)take((size_t)N * FDIM * 4);
  float*    ebuf   = (float*)take((size_t)Et * 4 * 4);
  float*    alS    = (float*)take((size_t)N * 4 * 4);
  float*    alD    = (float*)take((size_t)N * 4 * 4);
  float*    mbuf   = (float*)take((size_t)N * 4 * 4);
  float*    sbuf   = (float*)take((size_t)N * 4 * 4);
  _Float16* Bp     = (_Float16*)take((size_t)FDIM * FDIM * 2);
  float*    pooled = (float*)take((size_t)G * FDIM * 4);
  float*    cnt    = (float*)take((size_t)G * 4);

  auto blocks = [](long n) { return dim3((unsigned)((n + THREADS - 1) / THREADS)); };

  // h16 <- f16(emb[x])  [N x 64]
  k_embed<<<blocks(N * 64), THREADS, 0, stream>>>(x, emb, h16, (int)N);

  for (int layer = 0; layer < 2; ++layer) {
    const float* W  = layer ? W2  : W1;
    const float* as = layer ? as2 : as1;
    const float* ad = layer ? ad2 : ad1;
    const float* bb = layer ? b2  : b1;
    const int K = layer ? FDIM : 64;

    k_pack_b<<<blocks((long)K * FDIM), THREADS, 0, stream>>>(W, Bp, K, FDIM);

    const long mtiles  = (N + 15) / 16;
    const long ngroups = (FDIM / 16) / NBLK;
    const long waves   = mtiles * ngroups;
    k_gemm_wmma<<<blocks(waves * 32), THREADS, 0, stream>>>(h16, Bp, hg, (int)N, K, FDIM);

    k_alpha<<<blocks(N * 4), THREADS, 0, stream>>>(hg, as, ad, alS, alD, (int)N);

    k_fill_f32<<<blocks(N * FDIM), THREADS, 0, stream>>>(agg, 0.f, N * FDIM);
    k_fill_f32<<<blocks(N * 4), THREADS, 0, stream>>>(mbuf, -3.402823466e+38f, N * 4);
    k_fill_f32<<<blocks(N * 4), THREADS, 0, stream>>>(sbuf, 0.f, N * 4);

    k_edge_max<<<blocks(Et), THREADS, 0, stream>>>(srcIdx, dstIdx, E, Et, alS, alD, ebuf, mbuf);
    k_edge_expsum<<<blocks(Et), THREADS, 0, stream>>>(srcIdx, dstIdx, E, Et, mbuf, ebuf, sbuf);
    k_edge_agg<<<blocks(Et * 4), THREADS, 0, stream>>>(srcIdx, dstIdx, E, Et, ebuf, sbuf, hg, agg);

    // hg <- relu(agg + b); h16 <- f16(hg) (input to next GEMM)
    k_bias_relu<<<blocks(N * FDIM), THREADS, 0, stream>>>(agg, bb, hg, h16, (int)N);
  }

  // global mean pool + linear head
  k_fill_f32<<<blocks((long)G * FDIM), THREADS, 0, stream>>>(pooled, 0.f, (long)G * FDIM);
  k_fill_f32<<<blocks(G), THREADS, 0, stream>>>(cnt, 0.f, G);
  k_pool<<<blocks(N), THREADS, 0, stream>>>(hg, batch, pooled, cnt, (int)N);
  k_final<<<blocks((long)G * 2), THREADS, 0, stream>>>(pooled, cnt, linW, linb, out, G);
}
